// Decoder_35347580846616
// MI455X (gfx1250) — matrix operand order repaired and checked by reading, not compiled, not run on hardware
//
#include <hip/hip_runtime.h>
#include <stdint.h>

// Problem constants (from reference)
#define NB   4
#define NN   2048
#define BN   (NB*NN)      // 8192 nodes
#define KK   30
#define FEAT 148          // 3*16 vec + 100 scalars

typedef __attribute__((ext_vector_type(16))) __bf16   v16bf;
typedef __attribute__((ext_vector_type(8)))  float    v8f;
typedef __attribute__((ext_vector_type(8)))  uint32_t v8u;
typedef __attribute__((ext_vector_type(4)))  uint32_t v4u;

__device__ __forceinline__ uint32_t f2bf(float f){
  uint32_t x = __builtin_bit_cast(uint32_t, f);
  return (x + 0x7FFFu + ((x >> 16) & 1u)) >> 16;   // RNE fp32 -> bf16
}
__device__ __forceinline__ uint32_t pack2(float lo, float hi){
  return f2bf(lo) | (f2bf(hi) << 16);
}

// 16-row GEMM tile.
// A: LDS, packed bf16 pairs, [16 rows x ldp dwords]; dword p = elements (2p, 2p+1).
//    Kpad = 32*ktiles (pairs 16*ktiles), zero-padded. ldp must be a multiple of 4.
// W: pre-packed bf16 B-fragments (frag index = nt*ktiles + kt, 256 dwords each).
// D: fp32 LDS [16 x ldd], writes cols [0, 16*ntiles).
__device__ __forceinline__ void gemm16(const uint32_t* Ap, int ldp,
                                       const uint32_t* wp, int ktiles, int ntiles,
                                       float* D, int ldd, int lane){
  const int row = lane & 15;
  const int hi4 = (lane >> 4) << 2;   // 0 or 4 (pair offset for upper half-wave)
  const int col = lane & 15;
  const uint32_t* arow = Ap + row * ldp + hi4;
  for (int nt = 0; nt < ntiles; ++nt){
    v8f acc = {0.f,0.f,0.f,0.f,0.f,0.f,0.f,0.f};
    const uint32_t* wnt = wp + (size_t)nt * ktiles * 256;
    for (int kt = 0; kt < ktiles; ++kt){
      const uint32_t* ap = arow + kt * 16;
      v4u l0 = *(const v4u*)(ap);         // ds_load_b128
      v4u l1 = *(const v4u*)(ap + 8);     // ds_load_b128
      v8u au = __builtin_shufflevector(l0, l1, 0,1,2,3,4,5,6,7);
      v16bf a  = __builtin_bit_cast(v16bf, au);
      v8u  bu  = *(const v8u*)(wnt + (size_t)kt * 256 + lane * 8);
      v16bf bm = __builtin_bit_cast(v16bf, bu);
      acc = __builtin_amdgcn_wmma_f32_16x16x32_bf16(false, a, false, bm,
                                                    (short)0, acc, false, false);
    }
    float* dp = D + (size_t)((lane >> 4) << 3) * ldd + nt * 16 + col;
#pragma unroll
    for (int r = 0; r < 8; ++r) dp[(size_t)r * ldd] = acc[r];
  }
}

// ---------------- weight packing: fp32 (din x dout, row-major) -> bf16 B-fragments
__global__ void pack_w(const float* __restrict__ W, int din, int dout,
                       int ktiles, int ntiles, uint32_t* __restrict__ out){
  int total = ktiles * ntiles * 256;
  for (int i = blockIdx.x * blockDim.x + threadIdx.x; i < total;
       i += gridDim.x * blockDim.x){
    int v     = i & 7;
    int lane  = (i >> 3) & 31;
    int frag  = i >> 8;
    int kt    = frag % ktiles;
    int nt    = frag / ktiles;
    int n     = nt * 16 + (lane & 15);
    int khalf = lane >> 4;
    int k0    = kt * 32 + (v >> 2) * 16 + khalf * 8 + (v & 3) * 2;
    float lo = (k0     < din && n < dout) ? W[(size_t)k0 * dout + n]       : 0.f;
    float hi = (k0 + 1 < din && n < dout) ? W[(size_t)(k0 + 1) * dout + n] : 0.f;
    out[i] = pack2(lo, hi);
  }
}

// ---------------- per-edge message kernel: one wave handles one node's 30 edges
struct EdgeArgs {
  const float *hv, *hv0, *hS, *hE;
  const int   *eidx;
  const float *mask;
  const uint32_t *wh0,*ws0,*wv0,*wh1,*ws1,*wv1,*wh2,*ws2,*wv2;
  const float *b0,*b1,*b2;
  float* meanOut;
};

__global__ void __launch_bounds__(32)
edge_kernel(EdgeArgs A){
  // packed bf16-pair A operands
  __shared__ uint32_t sAp [16][160];   // GVP0 ws in (264 s + 33 vn, K=320) / GVP2 ws in (K=128)
  __shared__ uint32_t sBp [16][64];    // GVP1 ws in (100 + 16 vn, K=128)
  __shared__ uint32_t vInp[3][16][32]; // GVP0 v in (33, K=64)
  __shared__ uint32_t vHp [3][16][32]; // vh packed (K=64 for GVP0 wv, K=32 for GVP1/2)
  __shared__ uint32_t vOp [3][16][16]; // gated vo packed (K=32)
  // fp32 GEMM outputs
  __shared__ float vHf[3][16][48];
  __shared__ float sOf[16][112];
  __shared__ float vOf[3][16][16];
  __shared__ float accM[160];

  int lane = (int)threadIdx.x;
  int node = (int)blockIdx.x;
  int b = node >> 11;               // N = 2048
  int n = node & 2047;
  float maskv = A.mask[node];

  for (int i = lane; i < 16*160; i += 32) ((uint32_t*)sAp)[i] = 0u;
  for (int i = lane; i < 16*64;  i += 32) ((uint32_t*)sBp)[i] = 0u;
  for (int i = lane; i < 3*16*32; i += 32){ ((uint32_t*)vInp)[i] = 0u; ((uint32_t*)vHp)[i] = 0u; }
  for (int i = lane; i < 3*16*16; i += 32) ((uint32_t*)vOp)[i] = 0u;
  for (int i = lane; i < 160; i += 32) accM[i] = 0.f;
  __syncthreads();

  const float* hvn = A.hv + (size_t)node * FEAT;

  for (int pass = 0; pass < 2; ++pass){
    int rows = pass ? 14 : 16;
    // ---- build m (packed) for this tile of edges
    for (int e = 0; e < rows; ++e){
      int k = pass * 16 + e;
      int j = A.eidx[(size_t)node * KK + k];
      const float* hvj  = A.hv  + (size_t)(b * NN + j) * FEAT;
      const float* hv0j = A.hv0 + (size_t)(b * NN + j) * FEAT;
      const float* hsj  = A.hS  + (size_t)(b * NN + j) * 32;
      const float* he   = A.hE  + ((size_t)node * KK + k) * 35;
      float ar  = (j < n) ? 1.f : 0.f;
      float arc = maskv * ar;
      float enc = maskv * (1.f - ar);
      for (int p = lane; p < 132; p += 32){
        float lo, hi;
        if (p < 50){        int i = 2*p;        lo = hvn[48+i];      hi = hvn[49+i]; }
        else if (p < 66){   int i = 2*(p-50);   lo = maskv*he[3+i];  hi = maskv*he[4+i]; }
        else if (p < 82){   int i = 2*(p-66);   lo = arc*hsj[i];     hi = arc*hsj[i+1]; }
        else {              int i = 2*(p-82);
                            lo = arc*hvj[48+i] + enc*hv0j[48+i];
                            hi = arc*hvj[49+i] + enc*hv0j[49+i]; }
        sAp[e][p] = pack2(lo, hi);
      }
      for (int idx = lane; idx < 51; idx += 32){
        int t = idx / 17, p = idx % 17;
        int c0 = 2*p, c1 = c0 + 1;
        float lo = (c0 < 16) ? hvn[t*16+c0]
                 : (c0 == 16 ? maskv*he[t]
                             : arc*hvj[t*16+c0-17] + enc*hv0j[t*16+c0-17]);
        float hi = (c1 < 16) ? hvn[t*16+c1]
                 : (c1 == 16 ? maskv*he[t]
                 : (c1 < 33 ? arc*hvj[t*16+c1-17] + enc*hv0j[t*16+c1-17] : 0.f));
        vInp[t][e][p] = pack2(lo, hi);
      }
    }
    __syncthreads();

    // ---- GVP0 -------------------------------------------------------------
    for (int t = 0; t < 3; ++t) gemm16(&vInp[t][0][0], 32, A.wh0, 2, 3, &vHf[t][0][0], 48, lane);
    __syncthreads();
    for (int idx = lane; idx < 16*17; idx += 32){   // vn (33) -> sAp pairs 132..148
      int e = idx / 17, p = idx % 17;
      int h0 = 2*p, h1 = h0 + 1;
      float s0 = vHf[0][e][h0]*vHf[0][e][h0] + vHf[1][e][h0]*vHf[1][e][h0] + vHf[2][e][h0]*vHf[2][e][h0];
      float lo = sqrtf(s0 + 1e-8f);
      float hi = 0.f;
      if (h1 < 33){
        float s1 = vHf[0][e][h1]*vHf[0][e][h1] + vHf[1][e][h1]*vHf[1][e][h1] + vHf[2][e][h1]*vHf[2][e][h1];
        hi = sqrtf(s1 + 1e-8f);
      }
      sAp[e][132 + p] = pack2(lo, hi);
    }
    for (int idx = lane; idx < 3*16*24; idx += 32){ // pack vh -> vHp pairs 0..23 (cols 33..47 are 0)
      int t = idx / 384, r = idx % 384, e = r / 24, p = r % 24;
      vHp[t][e][p] = pack2(vHf[t][e][2*p], vHf[t][e][2*p+1]);
    }
    __syncthreads();
    gemm16(&sAp[0][0], 160, A.ws0, 10, 7, &sOf[0][0], 112, lane);
    __syncthreads();
    for (int idx = lane; idx < 16*50; idx += 32){   // bias+relu -> sBp pairs 0..49
      int e = idx / 50, p = idx % 50; int i = 2*p;
      float lo = sOf[e][i]   + A.b0[i];   lo = lo > 0.f ? lo : 0.f;
      float hi = sOf[e][i+1] + A.b0[i+1]; hi = hi > 0.f ? hi : 0.f;
      sBp[e][p] = pack2(lo, hi);
    }
    for (int t = 0; t < 3; ++t) gemm16(&vHp[t][0][0], 32, A.wv0, 2, 1, &vOf[t][0][0], 16, lane);
    __syncthreads();
    for (int idx = lane; idx < 16*8; idx += 32){    // gate -> vOp
      int e = idx / 8, p = idx % 8; int c0 = 2*p, c1 = c0 + 1;
      float s0 = vOf[0][e][c0]*vOf[0][e][c0] + vOf[1][e][c0]*vOf[1][e][c0] + vOf[2][e][c0]*vOf[2][e][c0];
      float s1 = vOf[0][e][c1]*vOf[0][e][c1] + vOf[1][e][c1]*vOf[1][e][c1] + vOf[2][e][c1]*vOf[2][e][c1];
      float g0 = 1.f / (1.f + expf(-sqrtf(s0 + 1e-8f)));
      float g1 = 1.f / (1.f + expf(-sqrtf(s1 + 1e-8f)));
      for (int t = 0; t < 3; ++t)
        vOp[t][e][p] = pack2(vOf[t][e][c0]*g0, vOf[t][e][c1]*g1);
    }
    __syncthreads();

    // ---- GVP1 -------------------------------------------------------------
    for (int t = 0; t < 3; ++t) gemm16(&vOp[t][0][0], 16, A.wh1, 1, 1, &vHf[t][0][0], 48, lane);
    __syncthreads();
    for (int idx = lane; idx < 16*8; idx += 32){    // vn1 -> sBp pairs 50..57, pack vh
      int e = idx / 8, p = idx % 8; int h0 = 2*p, h1 = h0 + 1;
      float s0 = vHf[0][e][h0]*vHf[0][e][h0] + vHf[1][e][h0]*vHf[1][e][h0] + vHf[2][e][h0]*vHf[2][e][h0];
      float s1 = vHf[0][e][h1]*vHf[0][e][h1] + vHf[1][e][h1]*vHf[1][e][h1] + vHf[2][e][h1]*vHf[2][e][h1];
      sBp[e][50 + p] = pack2(sqrtf(s0 + 1e-8f), sqrtf(s1 + 1e-8f));
      for (int t = 0; t < 3; ++t)
        vHp[t][e][p] = pack2(vHf[t][e][h0], vHf[t][e][h1]);
    }
    for (int idx = lane; idx < 3*16*9; idx += 32){  // zero vHp pairs 8..16
      int t = idx / 144, r = idx % 144, e = r / 9, p = r % 9;
      vHp[t][e][8 + p] = 0u;
    }
    __syncthreads();
    gemm16(&sBp[0][0], 64, A.ws1, 4, 7, &sOf[0][0], 112, lane);
    __syncthreads();
    for (int idx = lane; idx < 16*50; idx += 32){   // bias+relu -> sAp pairs 0..49
      int e = idx / 50, p = idx % 50; int i = 2*p;
      float lo = sOf[e][i]   + A.b1[i];   lo = lo > 0.f ? lo : 0.f;
      float hi = sOf[e][i+1] + A.b1[i+1]; hi = hi > 0.f ? hi : 0.f;
      sAp[e][p] = pack2(lo, hi);
    }
    for (int idx = lane; idx < 16*6; idx += 32){    // zero sAp pairs 58..63 (K=128 pad)
      int e = idx / 6; sAp[e][58 + idx % 6] = 0u;
    }
    for (int t = 0; t < 3; ++t) gemm16(&vHp[t][0][0], 32, A.wv1, 1, 1, &vOf[t][0][0], 16, lane);
    __syncthreads();
    for (int idx = lane; idx < 16*8; idx += 32){    // gate -> vOp
      int e = idx / 8, p = idx % 8; int c0 = 2*p, c1 = c0 + 1;
      float s0 = vOf[0][e][c0]*vOf[0][e][c0] + vOf[1][e][c0]*vOf[1][e][c0] + vOf[2][e][c0]*vOf[2][e][c0];
      float s1 = vOf[0][e][c1]*vOf[0][e][c1] + vOf[1][e][c1]*vOf[1][e][c1] + vOf[2][e][c1]*vOf[2][e][c1];
      float g0 = 1.f / (1.f + expf(-sqrtf(s0 + 1e-8f)));
      float g1 = 1.f / (1.f + expf(-sqrtf(s1 + 1e-8f)));
      for (int t = 0; t < 3; ++t)
        vOp[t][e][p] = pack2(vOf[t][e][c0]*g0, vOf[t][e][c1]*g1);
    }
    __syncthreads();

    // ---- GVP2 (no relu / no gate) ----------------------------------------
    for (int t = 0; t < 3; ++t) gemm16(&vOp[t][0][0], 16, A.wh2, 1, 1, &vHf[t][0][0], 48, lane);
    __syncthreads();
    for (int idx = lane; idx < 16*8; idx += 32){    // vn2 -> sAp pairs 50..57, pack vh
      int e = idx / 8, p = idx % 8; int h0 = 2*p, h1 = h0 + 1;
      float s0 = vHf[0][e][h0]*vHf[0][e][h0] + vHf[1][e][h0]*vHf[1][e][h0] + vHf[2][e][h0]*vHf[2][e][h0];
      float s1 = vHf[0][e][h1]*vHf[0][e][h1] + vHf[1][e][h1]*vHf[1][e][h1] + vHf[2][e][h1]*vHf[2][e][h1];
      sAp[e][50 + p] = pack2(sqrtf(s0 + 1e-8f), sqrtf(s1 + 1e-8f));
      for (int t = 0; t < 3; ++t)
        vHp[t][e][p] = pack2(vHf[t][e][h0], vHf[t][e][h1]);
    }
    __syncthreads();
    gemm16(&sAp[0][0], 160, A.ws2, 4, 7, &sOf[0][0], 112, lane);
    __syncthreads();
    for (int idx = lane; idx < 1600; idx += 32){    // bias only (final scalars, fp32)
      int e = idx / 100, i = idx % 100; sOf[e][i] += A.b2[i];
    }
    for (int t = 0; t < 3; ++t) gemm16(&vHp[t][0][0], 32, A.wv2, 1, 1, &vOf[t][0][0], 16, lane);
    __syncthreads();

    // ---- accumulate masked mean over valid rows
    for (int f = lane; f < FEAT; f += 32){
      float s = 0.f;
      for (int r = 0; r < rows; ++r)
        s += (f < 48) ? vOf[f >> 4][r][f & 15] : sOf[r][f - 48];
      accM[f] += s;
    }
    __syncthreads();
  }
  for (int f = lane; f < FEAT; f += 32)
    A.meanOut[(size_t)node * FEAT + f] = accM[f] * (1.f / 30.f);
}

// ---------------- node update kernel: one wave handles 16 nodes
struct NodeArgs {
  const float *hv, *meanM, *mask;
  const uint32_t *whA,*wsA,*wvA,*whB,*wsB,*wvB;
  const float *bA,*bB,*ln0b,*ln0g,*ln1b,*ln1g;
  float *hRes, *out;
};

__global__ void __launch_bounds__(32)
node_kernel(NodeArgs P){
  __shared__ uint32_t sInp [16][80];    // ln0 s + vn0 (132, K=160)
  __shared__ uint32_t sMidp[16][224];   // dh0 out + vn1 (432, K=448)
  __shared__ uint32_t vIp  [3][16][16]; // ln0 v (K=32) -> reused for gated vo (K=32)
  __shared__ uint32_t vHp  [3][16][16]; // packed vh (K=32)
  __shared__ float vHf[3][16][32];
  __shared__ float sOf[16][400];        // raw h stage / ws outputs (fp32)
  __shared__ float vOf[3][16][32];
  __shared__ float red[16][3];

  int lane = (int)threadIdx.x;
  int node0 = (int)blockIdx.x * 16;

  for (int i = lane; i < 16*80;   i += 32) ((uint32_t*)sInp)[i]  = 0u;
  for (int i = lane; i < 16*224;  i += 32) ((uint32_t*)sMidp)[i] = 0u;
  for (int i = lane; i < 3*16*16; i += 32){ ((uint32_t*)vIp)[i] = 0u; ((uint32_t*)vHp)[i] = 0u; }
  __syncthreads();

  // stage raw h = h_V + mean_m into sOf cols 0..147
  for (int e = 0; e < 16; ++e){
    const float* hv = P.hv    + (size_t)(node0 + e) * FEAT;
    const float* mm = P.meanM + (size_t)(node0 + e) * FEAT;
    for (int i = lane; i < FEAT; i += 32) sOf[e][i] = hv[i] + mm[i];
  }
  __syncthreads();

  // ln0 reductions (one lane per node)
  if (lane < 16){
    int e = lane;
    float sv = 0.f;
    for (int i = 0; i < 48; ++i){ float x = sOf[e][i]; sv += x * x; }
    float ss = 0.f; for (int i = 0; i < 100; ++i) ss += sOf[e][48 + i];
    float mu = ss * 0.01f;
    float vs = 0.f; for (int i = 0; i < 100; ++i){ float x = sOf[e][48 + i] - mu; vs += x * x; }
    red[e][0] = sqrtf(sv * (1.f / 16.f) + 1e-8f);
    red[e][1] = mu;
    red[e][2] = 1.f / sqrtf(vs * 0.01f + 1e-3f);
  }
  __syncthreads();

  // normalize -> packed inputs + residual (global)
  for (int e = 0; e < 16; ++e){
    float vn = red[e][0], mu = red[e][1], rs = red[e][2];
    float* hres = P.hRes + (size_t)(node0 + e) * FEAT;
    for (int p = lane; p < 50; p += 32){
      int i = 2*p;
      float lo = (sOf[e][48+i]   - mu) * rs * P.ln0g[i]   + P.ln0b[i];
      float hi = (sOf[e][49+i]   - mu) * rs * P.ln0g[i+1] + P.ln0b[i+1];
      sInp[e][p] = pack2(lo, hi);
      hres[48+i] = lo; hres[49+i] = hi;
    }
    for (int idx = lane; idx < 24; idx += 32){
      int t = idx / 8, p = idx % 8; int c = t*16 + 2*p;
      float lo = sOf[e][c] / vn, hi = sOf[e][c+1] / vn;
      vIp[t][e][p] = pack2(lo, hi);
      hres[c] = lo; hres[c+1] = hi;
    }
  }
  __syncthreads();

  // ---- dh0 (relu + gate)
  for (int t = 0; t < 3; ++t) gemm16(&vIp[t][0][0], 16, P.whA, 1, 2, &vHf[t][0][0], 32, lane);
  __syncthreads();
  for (int idx = lane; idx < 16*16; idx += 32){   // vn0 (32) -> sInp pairs 50..65, pack vh
    int e = idx / 16, p = idx % 16; int h0 = 2*p, h1 = h0 + 1;
    float s0 = vHf[0][e][h0]*vHf[0][e][h0] + vHf[1][e][h0]*vHf[1][e][h0] + vHf[2][e][h0]*vHf[2][e][h0];
    float s1 = vHf[0][e][h1]*vHf[0][e][h1] + vHf[1][e][h1]*vHf[1][e][h1] + vHf[2][e][h1]*vHf[2][e][h1];
    sInp[e][50 + p] = pack2(sqrtf(s0 + 1e-8f), sqrtf(s1 + 1e-8f));
    for (int t = 0; t < 3; ++t)
      vHp[t][e][p] = pack2(vHf[t][e][h0], vHf[t][e][h1]);
  }
  __syncthreads();
  gemm16(&sInp[0][0], 80, P.wsA, 5, 25, &sOf[0][0], 400, lane);
  __syncthreads();
  for (int idx = lane; idx < 16*200; idx += 32){  // bias+relu -> sMidp pairs 0..199
    int e = idx / 200, p = idx % 200; int i = 2*p;
    float lo = sOf[e][i]   + P.bA[i];   lo = lo > 0.f ? lo : 0.f;
    float hi = sOf[e][i+1] + P.bA[i+1]; hi = hi > 0.f ? hi : 0.f;
    sMidp[e][p] = pack2(lo, hi);
  }
  for (int t = 0; t < 3; ++t) gemm16(&vHp[t][0][0], 16, P.wvA, 1, 2, &vOf[t][0][0], 32, lane);
  __syncthreads();
  for (int idx = lane; idx < 16*16; idx += 32){   // gate -> vIp (reused, K=32 exact)
    int e = idx / 16, p = idx % 16; int c0 = 2*p, c1 = c0 + 1;
    float s0 = vOf[0][e][c0]*vOf[0][e][c0] + vOf[1][e][c0]*vOf[1][e][c0] + vOf[2][e][c0]*vOf[2][e][c0];
    float s1 = vOf[0][e][c1]*vOf[0][e][c1] + vOf[1][e][c1]*vOf[1][e][c1] + vOf[2][e][c1]*vOf[2][e][c1];
    float g0 = 1.f / (1.f + expf(-sqrtf(s0 + 1e-8f)));
    float g1 = 1.f / (1.f + expf(-sqrtf(s1 + 1e-8f)));
    for (int t = 0; t < 3; ++t)
      vIp[t][e][p] = pack2(vOf[t][e][c0]*g0, vOf[t][e][c1]*g1);
  }
  __syncthreads();

  // ---- dh1 (linear)
  for (int t = 0; t < 3; ++t) gemm16(&vIp[t][0][0], 16, P.whB, 1, 2, &vHf[t][0][0], 32, lane);
  __syncthreads();
  for (int idx = lane; idx < 16*16; idx += 32){   // vn1 (32) -> sMidp pairs 200..215, pack vh
    int e = idx / 16, p = idx % 16; int h0 = 2*p, h1 = h0 + 1;
    float s0 = vHf[0][e][h0]*vHf[0][e][h0] + vHf[1][e][h0]*vHf[1][e][h0] + vHf[2][e][h0]*vHf[2][e][h0];
    float s1 = vHf[0][e][h1]*vHf[0][e][h1] + vHf[1][e][h1]*vHf[1][e][h1] + vHf[2][e][h1]*vHf[2][e][h1];
    sMidp[e][200 + p] = pack2(sqrtf(s0 + 1e-8f), sqrtf(s1 + 1e-8f));
    for (int t = 0; t < 3; ++t)
      vHp[t][e][p] = pack2(vHf[t][e][h0], vHf[t][e][h1]);
  }
  __syncthreads();
  gemm16(&sMidp[0][0], 224, P.wsB, 14, 7, &sOf[0][0], 400, lane);
  __syncthreads();
  for (int idx = lane; idx < 1600; idx += 32){    // bias only (fp32)
    int e = idx / 100, i = idx % 100; sOf[e][i] += P.bB[i];
  }
  for (int t = 0; t < 3; ++t) gemm16(&vHp[t][0][0], 16, P.wvB, 1, 1, &vOf[t][0][0], 32, lane);
  __syncthreads();

  // ---- ln1(residual + d) * mask
  if (lane < 16){
    int e = lane, node = node0 + e;
    const float* hr = P.hRes + (size_t)node * FEAT;
    float maskv = P.mask[node];
    float sv = 0.f;
    for (int idx = 0; idx < 48; ++idx){
      float x = hr[idx] + vOf[idx >> 4][e][idx & 15]; sv += x * x;
    }
    float ss = 0.f; for (int i = 0; i < 100; ++i) ss += hr[48 + i] + sOf[e][i];
    float mu = ss * 0.01f;
    float vs = 0.f;
    for (int i = 0; i < 100; ++i){ float x = hr[48 + i] + sOf[e][i] - mu; vs += x * x; }
    float vn = sqrtf(sv * (1.f / 16.f) + 1e-8f);
    float rs = 1.f / sqrtf(vs * 0.01f + 1e-3f);
    float* out = P.out + (size_t)node * FEAT;
    for (int idx = 0; idx < 48; ++idx)
      out[idx] = maskv * (hr[idx] + vOf[idx >> 4][e][idx & 15]) / vn;
    for (int i = 0; i < 100; ++i){
      float x = hr[48 + i] + sOf[e][i];
      out[48 + i] = maskv * ((x - mu) * rs * P.ln1g[i] + P.ln1b[i]);
    }
  }
}

// ----------------------------- host side -----------------------------------
extern "C" void kernel_launch(void* const* d_in, const int* in_sizes, int n_in,
                              void* d_out, int out_size, void* d_ws, size_t ws_size,
                              hipStream_t stream){
  (void)in_sizes; (void)n_in; (void)out_size; (void)ws_size;
  const float* hV0  = (const float*)d_in[0];
  const float* hS   = (const float*)d_in[1];
  const float* hE   = (const float*)d_in[2];
  const int*   Eidx = (const int*)  d_in[3];
  const float* mask = (const float*)d_in[4];

  // params pytree (per layer, 24 leaves; dict keys sorted):
  // W_EV[0..2]{wh,ws_b,ws_w,wv}, W_dh[0..1]{...}, ln0{b,g}, ln1{b,g}
  auto leaf = [&](int l, int off){ return (const float*)d_in[5 + l * 24 + off]; };

  struct MD { int leaf, din, dout, kt, nt; };
  static const MD md[15] = {
    { 0,  33,  33,  2,  3}, { 2, 297, 100, 10,  7}, { 3,  33,  16,  2,  1},  // EV0
    { 4,  16,  16,  1,  1}, { 6, 116, 100,  4,  7}, { 7,  16,  16,  1,  1},  // EV1
    { 8,  16,  16,  1,  1}, {10, 116, 100,  4,  7}, {11,  16,  16,  1,  1},  // EV2
    {12,  16,  32,  1,  2}, {14, 132, 400,  5, 25}, {15,  32,  32,  1,  2},  // dh0
    {16,  32,  32,  1,  2}, {18, 432, 100, 14,  7}, {19,  32,  16,  1,  1},  // dh1
  };

  uint32_t* wpack = (uint32_t*)d_ws;
  size_t off[3][15];
  size_t dw = 0;
  for (int l = 0; l < 3; ++l)
    for (int i = 0; i < 15; ++i){
      off[l][i] = dw;
      int total = md[i].kt * md[i].nt * 256;
      pack_w<<<(total + 255) / 256, 256, 0, stream>>>(
          leaf(l, md[i].leaf), md[i].din, md[i].dout, md[i].kt, md[i].nt,
          wpack + dw);
      dw += (size_t)total;
    }

  float* meanM = (float*)((char*)d_ws + ((dw * 4 + 255) & ~(size_t)255));
  float* hRes  = meanM + (size_t)BN * FEAT;
  float* hvA   = hRes  + (size_t)BN * FEAT;
  float* hvB   = hvA   + (size_t)BN * FEAT;

  for (int l = 0; l < 3; ++l){
    const float* hv_cur = (l == 0) ? hV0 : ((l == 1) ? hvA : hvB);
    float* hv_out = (l == 2) ? (float*)d_out : ((l == 0) ? hvA : hvB);

    EdgeArgs ea;
    ea.hv = hv_cur; ea.hv0 = hV0; ea.hS = hS; ea.hE = hE;
    ea.eidx = Eidx; ea.mask = mask;
    ea.wh0 = wpack + off[l][0];  ea.ws0 = wpack + off[l][1];  ea.wv0 = wpack + off[l][2];
    ea.wh1 = wpack + off[l][3];  ea.ws1 = wpack + off[l][4];  ea.wv1 = wpack + off[l][5];
    ea.wh2 = wpack + off[l][6];  ea.ws2 = wpack + off[l][7];  ea.wv2 = wpack + off[l][8];
    ea.b0 = leaf(l, 1); ea.b1 = leaf(l, 5); ea.b2 = leaf(l, 9);
    ea.meanOut = meanM;
    edge_kernel<<<BN, 32, 0, stream>>>(ea);

    NodeArgs na;
    na.hv = hv_cur; na.meanM = meanM; na.mask = mask;
    na.whA = wpack + off[l][9];  na.wsA = wpack + off[l][10]; na.wvA = wpack + off[l][11];
    na.whB = wpack + off[l][12]; na.wsB = wpack + off[l][13]; na.wvB = wpack + off[l][14];
    na.bA = leaf(l, 13); na.bB = leaf(l, 17);
    na.ln0b = leaf(l, 20); na.ln0g = leaf(l, 21);
    na.ln1b = leaf(l, 22); na.ln1g = leaf(l, 23);
    na.hRes = hRes; na.out = hv_out;
    node_kernel<<<BN / 16, 32, 0, stream>>>(na);
  }
}
